// NeuralAdaptiveNetwork_82136954569340
// MI455X (gfx1250) — matrix operand order repaired and checked by reading, not compiled, not run on hardware
//
#include <hip/hip_runtime.h>
#include <hip/hip_bf16.h>
#include <math.h>

// ---------------------------------------------------------------------------
// MI455X / gfx1250 implementation.
// Heavy math: v_wmma_f32_16x16x32_bf16 (wave32 WMMA).
// LSTM tile staging: double-buffered TDM tensor_load_to_lds + s_wait_tensorcnt.
// ---------------------------------------------------------------------------

typedef __attribute__((ext_vector_type(16))) __bf16 v16bf;
typedef __attribute__((ext_vector_type(8)))  __bf16 v8bf;
typedef __attribute__((ext_vector_type(4)))  __bf16 v4bf;
typedef __attribute__((ext_vector_type(8)))  float  v8f;
typedef __attribute__((ext_vector_type(4)))  unsigned u32x4;
typedef __attribute__((ext_vector_type(8)))  int      i32x8;
typedef __attribute__((ext_vector_type(4)))  int      i32x4;

#define BDIM 256
#define LRS 40   // LDS row stride in bf16 elems (32 data + 8 pad; TDM pad cfg)
constexpr int Bc  = 64;
constexpr int Sc  = 512;
constexpr int INc = 1024;
constexpr int Hc  = 512;
constexpr int G4  = 2048;
constexpr float EPSc = 1e-5f;

#if defined(__has_builtin)
#if __has_builtin(__builtin_amdgcn_tensor_load_to_lds) && __has_builtin(__builtin_amdgcn_s_wait_tensorcnt)
#define HAVE_TDM 1
#endif
#endif
#ifndef HAVE_TDM
#define HAVE_TDM 0
#endif

#if HAVE_TDM
#pragma message("CDNA5: TDM tensor_load_to_lds path ENABLED")
#else
#warning "CDNA5: TDM builtin NOT available; using manual LDS staging"
#endif

__device__ __forceinline__ float sigm(float x) { return 1.0f / (1.0f + __expf(-x)); }

union bfu { v16bf v; v8bf h[2]; };
__device__ __forceinline__ v16bf load_frag2(const __bf16* p0, const __bf16* p1) {
  bfu u;
  u.h[0] = *(const v8bf*)p0;
  u.h[1] = *(const v8bf*)p1;
  return u.v;
}

__device__ __forceinline__ v8f wmma_bf16(v16bf a, v16bf b, v8f c) {
  return __builtin_amdgcn_wmma_f32_16x16x32_bf16(
      false, a, false, b, (short)0, c, false, false);
}

#if HAVE_TDM
__device__ __forceinline__ void tdm_issue(u32x4 g0, i32x8 g1, i32x4 g2, i32x4 g3) {
#if __clang_major__ >= 23
  i32x8 zx = {0, 0, 0, 0, 0, 0, 0, 0};
  __builtin_amdgcn_tensor_load_to_lds(g0, g1, g2, g3, zx, 0);
#else
  __builtin_amdgcn_tensor_load_to_lds(g0, g1, g2, g3, 0);
#endif
}

// Load a (tile_z x tile_rows x 32) bf16 tile into LDS, row-major with
// +8-elem row padding (pad_interval=16 dwords, pad_amount=4 dwords -> LRS=40).
__device__ __forceinline__ void tdm_tile_load(unsigned ldsoff, const __bf16* gp,
                                              unsigned tile_rows, unsigned tile_z,
                                              unsigned row_stride, unsigned long z_stride) {
  unsigned long long ga = (unsigned long long)(uintptr_t)gp;
  u32x4 g0; i32x8 g1; i32x4 g2, g3;
  g0[0] = 1u;                                   // count=1 (user descriptor)
  g0[1] = ldsoff;                               // lds_addr (bytes)
  g0[2] = (unsigned)ga;                         // global_addr[31:0]
  g0[3] = (unsigned)((ga >> 32) & 0x1ffffffu) | (2u << 30);  // addr[56:32] | type=2
  unsigned td0 = row_stride;                    // tensor_dim0 (safe: >= tile_k)
  unsigned td1 = tile_rows;
  // dw0: data_size=1(2B)@16, pad_enable@20, pad_interval=3(16dw)@22, pad_amount=3(4dw)@25
  g1[0] = (int)((1u << 16) | (1u << 20) | (3u << 22) | (3u << 25));
  g1[1] = (int)((td0 & 0xffffu) << 16);                         // tensor_dim0 lo16
  g1[2] = (int)((td0 >> 16) | ((td1 & 0xffffu) << 16));         // td0 hi16 | td1 lo16
  g1[3] = (int)((td1 >> 16) | (32u << 16));                     // td1 hi16 | tile_dim0=32
  g1[4] = (int)((tile_rows & 0xffffu) | (tile_z << 16));        // tile_dim1 | tile_dim2
  g1[5] = (int)row_stride;                                      // dim0_stride[31:0]
  g1[6] = (int)((unsigned)(z_stride & 0xffffu) << 16);          // d0s hi16=0 | d1s lo16
  g1[7] = (int)(z_stride >> 16);                                // dim1_stride[47:16]
  g2[0] = (int)tile_z; g2[1] = 0; g2[2] = 0; g2[3] = 0;         // tensor_dim2
  g3[0] = 0; g3[1] = 0; g3[2] = 0; g3[3] = 0;
  tdm_issue(g0, g1, g2, g3);
}

// A tile (64x32 of h-state) + W tile (4 gates x 64 rows x 32) as one pair.
__device__ __forceinline__ void issue_pair(unsigned offA, unsigned offW,
                                           const __bf16* aptr, const __bf16* wptr,
                                           int hs, int k0) {
  tdm_tile_load(offA, aptr + k0, 64, 0, Hc, 0);
  tdm_tile_load(offW, wptr + (size_t)hs * Hc + k0, 64, 4, Hc,
                (unsigned long)Hc * 512u);
}
#endif

// ---------------------------------------------------------------------------
// Generic WMMA GEMM: C(M,N) = act(A(M,K) @ W(N,K)^T + bias(N))
// f32 in/out; f32->bf16 cvt during LDS staging (float4 loads, v4bf stores).
// Block 256 thr (8 waves); tile BM=64, BN=128, BK=32; row-major LDS (LRS=40).
// ---------------------------------------------------------------------------
__global__ __launch_bounds__(BDIM)
void wmma_gemm_bias_act(const float* __restrict__ A, const float* __restrict__ W,
                        const float* __restrict__ bias, float* __restrict__ C,
                        int M, int N, int K, int act)
{
  __shared__ __align__(16) __bf16 ldsA[64 * LRS];
  __shared__ __align__(16) __bf16 ldsB[128 * LRS];
  const int tid  = threadIdx.x;
  const int wave = tid >> 5;
  const int lane = tid & 31;
  const int laneN = lane & 15, hiK = lane >> 4;
  const int bm = blockIdx.x * 64;
  const int bn = blockIdx.y * 128;

  v8f acc[4];
  #pragma unroll
  for (int i = 0; i < 4; ++i)
    #pragma unroll
    for (int r = 0; r < 8; ++r) acc[i][r] = 0.0f;

  for (int k0 = 0; k0 < K; k0 += 32) {
    if (k0 + 32 < K) {
      __builtin_prefetch(&A[(size_t)(bm + (tid & 63)) * K + k0 + 32], 0, 3);
      __builtin_prefetch(&W[(size_t)(bn + (tid & 127)) * K + k0 + 32], 0, 3);
    }
    __syncthreads();
    // A tile 64x32: 512 float4 chunks, 2/thread
    #pragma unroll
    for (int e = 0; e < 2; ++e) {
      int c = e * BDIM + tid;
      int m = c >> 3, k4 = c & 7;
      float4 f = *(const float4*)&A[(size_t)(bm + m) * K + k0 + k4 * 4];
      v4bf b4 = { (__bf16)f.x, (__bf16)f.y, (__bf16)f.z, (__bf16)f.w };
      *(v4bf*)&ldsA[m * LRS + k4 * 4] = b4;
    }
    // B tile 128x32: 1024 chunks, 4/thread
    #pragma unroll
    for (int e = 0; e < 4; ++e) {
      int c = e * BDIM + tid;
      int n = c >> 3, k4 = c & 7;
      float4 f = *(const float4*)&W[(size_t)(bn + n) * K + k0 + k4 * 4];
      v4bf b4 = { (__bf16)f.x, (__bf16)f.y, (__bf16)f.z, (__bf16)f.w };
      *(v4bf*)&ldsB[n * LRS + k4 * 4] = b4;
    }
    __syncthreads();
    const __bf16* br = &ldsB[(wave * 16 + laneN) * LRS + 16 * hiK];
    v16bf bf_ = load_frag2(br, br + 8);
    #pragma unroll
    for (int mi = 0; mi < 4; ++mi) {
      const __bf16* ar = &ldsA[(mi * 16 + laneN) * LRS + 16 * hiK];
      acc[mi] = wmma_bf16(load_frag2(ar, ar + 8), bf_, acc[mi]);
    }
  }

  int col = bn + (wave << 4) + laneN;
  float bv = bias ? bias[col] : 0.0f;
  #pragma unroll
  for (int mi = 0; mi < 4; ++mi)
    #pragma unroll
    for (int r = 0; r < 8; ++r) {
      int row = bm + mi * 16 + (hiK << 3) + r;
      float v = acc[mi][r] + bv;
      if (act) v = fmaxf(v, 0.0f);
      C[(size_t)row * N + col] = v;
    }
}

// ---------------------------------------------------------------------------
// BatchNorm1d (channel = S) stats + apply(ReLU)
// ---------------------------------------------------------------------------
__global__ __launch_bounds__(BDIM)
void bn_stats(const float* __restrict__ h, float* __restrict__ mean,
              float* __restrict__ rvar)
{
  int s = blockIdx.x, tid = threadIdx.x;
  __shared__ float r1[BDIM], r2[BDIM];
  float sum = 0.f, sq = 0.f;
  for (int i = tid; i < Bc * Hc; i += BDIM) {
    int b = i >> 9, j = i & 511;
    float v = h[((size_t)(b * Sc + s)) * Hc + j];
    sum += v; sq += v * v;
  }
  r1[tid] = sum; r2[tid] = sq; __syncthreads();
  for (int off = BDIM / 2; off > 0; off >>= 1) {
    if (tid < off) { r1[tid] += r1[tid + off]; r2[tid] += r2[tid + off]; }
    __syncthreads();
  }
  if (tid == 0) {
    float m = r1[0] / (float)(Bc * Hc);
    float v = r2[0] / (float)(Bc * Hc) - m * m;
    mean[s] = m; rvar[s] = rsqrtf(v + EPSc);
  }
}

__global__ __launch_bounds__(BDIM)
void bn_apply_relu(float* __restrict__ h, const float* __restrict__ mean,
                   const float* __restrict__ rvar, const float* __restrict__ g,
                   const float* __restrict__ bt)
{
  const int total = Bc * Sc * Hc;
  for (int i = blockIdx.x * BDIM + threadIdx.x; i < total; i += gridDim.x * BDIM) {
    int row = i >> 9;
    int s = row & (Sc - 1);
    float v = (h[i] - mean[s]) * rvar[s] * g[s] + bt[s];
    h[i] = fmaxf(v, 0.0f);
  }
}

// ---------------------------------------------------------------------------
// Persistent 2-layer LSTM. 8 blocks, each owns a 64-wide H slice with all 4
// gates in-register; bf16 h-state ping-pong; double-buffered TDM staging.
// ---------------------------------------------------------------------------
__device__ __forceinline__ void gbar(int* bar, int i, int tid, int nblk) {
  __syncthreads();
  if (tid == 0) {
    __hip_atomic_fetch_add(&bar[i], 1, __ATOMIC_ACQ_REL, __HIP_MEMORY_SCOPE_AGENT);
    while (__hip_atomic_load(&bar[i], __ATOMIC_ACQUIRE, __HIP_MEMORY_SCOPE_AGENT) < nblk)
      __builtin_amdgcn_s_sleep(1);
  }
  __syncthreads();
  __threadfence();
}

__device__ __forceinline__ void wmma_tile_compute(const __bf16* lw, const __bf16* la,
                                                  v8f (&acc)[2][4], int hsl, int mg,
                                                  int laneN, int hiK) {
  #pragma unroll
  for (int g = 0; g < 4; ++g) {
    const __bf16* br = &lw[((g * 4 + hsl) * 16 + laneN) * LRS + 16 * hiK];
    v16bf bf_ = load_frag2(br, br + 8);
    #pragma unroll
    for (int mi = 0; mi < 2; ++mi) {
      const __bf16* ar = &la[((mg * 2 + mi) * 16 + laneN) * LRS + 16 * hiK];
      acc[mi][g] = wmma_bf16(load_frag2(ar, ar + 8), bf_, acc[mi][g]);
    }
  }
}

#if !HAVE_TDM
__device__ __forceinline__ void stage_manual(const __bf16* __restrict__ a,
                                             const __bf16* __restrict__ w,
                                             __bf16* ldsA, __bf16* ldsW,
                                             int hs, int k0, int tid) {
  #pragma unroll
  for (int e = 0; e < 2; ++e) {            // A tile 64x32
    int c = e * BDIM + tid;
    int m = c >> 3, k4 = c & 7;
    *(v4bf*)&ldsA[m * LRS + k4 * 4] = *(const v4bf*)&a[m * Hc + k0 + k4 * 4];
  }
  #pragma unroll
  for (int e = 0; e < 8; ++e) {            // W tile 256x32 (gate-strided rows)
    int c = e * BDIM + tid;
    int r = c >> 3, k4 = c & 7;
    int g = r >> 6, hh = r & 63;
    *(v4bf*)&ldsW[r * LRS + k4 * 4] =
        *(const v4bf*)&w[(size_t)(g * Hc + hs + hh) * Hc + k0 + k4 * 4];
  }
}
#endif

__global__ __launch_bounds__(BDIM)
void lstm_seq_kernel(const float* __restrict__ X0,      // (B*S, 2048): x@Wih0^T + b0s
                     const __bf16* __restrict__ Whh0,
                     const __bf16* __restrict__ Wih1,
                     const __bf16* __restrict__ Whh1,
                     const float* __restrict__ b1s,
                     __bf16* __restrict__ h0b,          // 2 slots (64,512) bf16
                     float* __restrict__ c0,
                     __bf16* __restrict__ h1b,          // 2 slots
                     float* __restrict__ c1,
                     float* __restrict__ h1f,           // final f32 h1
                     int* __restrict__ bar)
{
  __shared__ __align__(16) __bf16 ldsW[2][256 * LRS];
  __shared__ __align__(16) __bf16 ldsA[2][64 * LRS];
  const int tid  = threadIdx.x;
  const int wave = tid >> 5, lane = tid & 31;
  const int mg = wave >> 2, hsl = wave & 3;
  const int hs = blockIdx.x * 64;
  const int laneN = lane & 15, hiK = lane >> 4, laneM8 = hiK << 3;
  const int wv = (int)(__builtin_amdgcn_readfirstlane(threadIdx.x) >> 5);
#if HAVE_TDM
  const unsigned offA[2] = { (unsigned)(uintptr_t)&ldsA[0][0],
                             (unsigned)(uintptr_t)&ldsA[1][0] };
  const unsigned offW[2] = { (unsigned)(uintptr_t)&ldsW[0][0],
                             (unsigned)(uintptr_t)&ldsW[1][0] };
#endif

  for (int t = 0; t < Sc; ++t) {
    const int pr = t & 1, cu = (t + 1) & 1;
    const __bf16* h0prev = h0b + pr * Bc * Hc;
    __bf16*       h0cur  = h0b + cu * Bc * Hc;
    const __bf16* h1prev = h1b + pr * Bc * Hc;
    __bf16*       h1cur  = h1b + cu * Bc * Hc;

    // ------------- layer 0: z = X0[:,t,slice] + h0prev @ Whh0_slice^T -----
    v8f acc[2][4];
    #pragma unroll
    for (int mi = 0; mi < 2; ++mi)
      #pragma unroll
      for (int g = 0; g < 4; ++g)
        #pragma unroll
        for (int r = 0; r < 8; ++r) acc[mi][g][r] = 0.0f;

#if HAVE_TDM
    if (wv == 0) issue_pair(offA[0], offW[0], h0prev, Whh0, hs, 0);
    for (int ks = 0; ks < 16; ++ks) {
      int cb = ks & 1;
      if (wv == 0) {
        if (ks < 15) {
          issue_pair(offA[cb ^ 1], offW[cb ^ 1], h0prev, Whh0, hs, (ks + 1) * 32);
          __builtin_amdgcn_s_wait_tensorcnt(2);   // previous pair has landed
        } else {
          __builtin_amdgcn_s_wait_tensorcnt(0);
        }
      }
      __syncthreads();
      wmma_tile_compute(ldsW[cb], ldsA[cb], acc, hsl, mg, laneN, hiK);
      __syncthreads();
    }
#else
    for (int ks = 0; ks < 16; ++ks) {
      int k0 = ks * 32;
      __syncthreads();
      stage_manual(h0prev + k0, Whh0, ldsA[0], ldsW[0], hs, k0, tid);
      __syncthreads();
      wmma_tile_compute(ldsW[0], ldsA[0], acc, hsl, mg, laneN, hiK);
    }
    __syncthreads();
#endif
    #pragma unroll
    for (int mi = 0; mi < 2; ++mi)
      #pragma unroll
      for (int r = 0; r < 8; ++r) {
        int row  = mg * 32 + mi * 16 + laneM8 + r;
        int hcol = hs + hsl * 16 + laneN;
        const float* xrow = &X0[((size_t)row * Sc + t) * G4];
        float zi = acc[mi][0][r] + xrow[0 * Hc + hcol];
        float zf = acc[mi][1][r] + xrow[1 * Hc + hcol];
        float zg = acc[mi][2][r] + xrow[2 * Hc + hcol];
        float zo = acc[mi][3][r] + xrow[3 * Hc + hcol];
        float cp = c0[row * Hc + hcol];
        float cn = sigm(zf) * cp + sigm(zi) * tanhf(zg);
        c0[row * Hc + hcol]   = cn;
        h0cur[row * Hc + hcol] = (__bf16)(sigm(zo) * tanhf(cn));
      }
    gbar(bar, 2 * t + 0, tid, 8);

    // ------------- layer 1: z = h0cur@Wih1^T + h1prev@Whh1^T + b1s --------
    #pragma unroll
    for (int mi = 0; mi < 2; ++mi)
      #pragma unroll
      for (int g = 0; g < 4; ++g)
        #pragma unroll
        for (int r = 0; r < 8; ++r) acc[mi][g][r] = 0.0f;

#if HAVE_TDM
    if (wv == 0) issue_pair(offA[0], offW[0], h0cur, Wih1, hs, 0);
    for (int ks = 0; ks < 32; ++ks) {
      int cb = ks & 1;
      if (wv == 0) {
        if (ks < 31) {
          int kn = ks + 1;
          const __bf16* na = (kn < 16) ? h0cur : h1prev;
          const __bf16* nw = (kn < 16) ? Wih1 : Whh1;
          issue_pair(offA[cb ^ 1], offW[cb ^ 1], na, nw, hs, (kn & 15) * 32);
          __builtin_amdgcn_s_wait_tensorcnt(2);
        } else {
          __builtin_amdgcn_s_wait_tensorcnt(0);
        }
      }
      __syncthreads();
      wmma_tile_compute(ldsW[cb], ldsA[cb], acc, hsl, mg, laneN, hiK);
      __syncthreads();
    }
#else
    for (int ks = 0; ks < 32; ++ks) {
      int k0 = (ks & 15) * 32;
      const __bf16* aptr = (ks < 16) ? h0cur : h1prev;
      const __bf16* wptr = (ks < 16) ? Wih1 : Whh1;
      __syncthreads();
      stage_manual(aptr + k0, wptr, ldsA[0], ldsW[0], hs, k0, tid);
      __syncthreads();
      wmma_tile_compute(ldsW[0], ldsA[0], acc, hsl, mg, laneN, hiK);
    }
    __syncthreads();
#endif
    #pragma unroll
    for (int mi = 0; mi < 2; ++mi)
      #pragma unroll
      for (int r = 0; r < 8; ++r) {
        int row  = mg * 32 + mi * 16 + laneM8 + r;
        int hcol = hs + hsl * 16 + laneN;
        float zi = acc[mi][0][r] + b1s[0 * Hc + hcol];
        float zf = acc[mi][1][r] + b1s[1 * Hc + hcol];
        float zg = acc[mi][2][r] + b1s[2 * Hc + hcol];
        float zo = acc[mi][3][r] + b1s[3 * Hc + hcol];
        float cp = c1[row * Hc + hcol];
        float cn = sigm(zf) * cp + sigm(zi) * tanhf(zg);
        float hv = sigm(zo) * tanhf(cn);
        c1[row * Hc + hcol]    = cn;
        h1cur[row * Hc + hcol] = (__bf16)hv;
        h1f[row * Hc + hcol]   = hv;     // last step's write = final hidden
      }
    gbar(bar, 2 * t + 1, tid, 8);
  }
}

// ---------------------------------------------------------------------------
// Small helper kernels
// ---------------------------------------------------------------------------
__global__ void init_zero(float* fz, int nf, int* iz, int ni) {
  int stride = gridDim.x * blockDim.x;
  for (int k = blockIdx.x * blockDim.x + threadIdx.x; k < nf; k += stride) fz[k] = 0.0f;
  for (int k = blockIdx.x * blockDim.x + threadIdx.x; k < ni; k += stride) iz[k] = 0;
}

__global__ void cvt_bf16(const float* __restrict__ in, __bf16* __restrict__ out, int n) {
  int i = blockIdx.x * blockDim.x + threadIdx.x;
  if (i < n) out[i] = (__bf16)in[i];
}

__global__ void addvec(const float* a, const float* b, float* o, int n) {
  int i = blockIdx.x * blockDim.x + threadIdx.x;
  if (i < n) o[i] = a[i] + b[i];
}

__global__ void concat_kernel(const float* __restrict__ a, const float* __restrict__ b,
                              float* __restrict__ o) {
  int i = blockIdx.x * blockDim.x + threadIdx.x;
  if (i < Bc * 2 * Hc) {
    int row = i >> 10, col = i & 1023;
    o[i] = (col < Hc) ? a[row * Hc + col] : b[row * Hc + (col - Hc)];
  }
}

__global__ void mul_kernel(const float* a, const float* b, float* o, int n) {
  int i = blockIdx.x * blockDim.x + threadIdx.x;
  if (i < n) o[i] = a[i] * b[i];
}

__global__ __launch_bounds__(BDIM)
void ln_act(const float* __restrict__ X, const float* __restrict__ g,
            const float* __restrict__ b, float* __restrict__ Y, int D, int dosig)
{
  int row = blockIdx.x, tid = threadIdx.x;
  __shared__ float r1[BDIM], r2[BDIM];
  float s = 0.f, q = 0.f;
  for (int j = tid; j < D; j += BDIM) { float v = X[row * D + j]; s += v; q += v * v; }
  r1[tid] = s; r2[tid] = q; __syncthreads();
  for (int off = BDIM / 2; off > 0; off >>= 1) {
    if (tid < off) { r1[tid] += r1[tid + off]; r2[tid] += r2[tid + off]; }
    __syncthreads();
  }
  float m = r1[0] / (float)D;
  float var = r2[0] / (float)D - m * m;
  float rv = rsqrtf(var + EPSc);
  for (int j = tid; j < D; j += BDIM) {
    float v = (X[row * D + j] - m) * rv * g[j] + b[j];
    if (dosig) v = sigm(v);
    Y[row * D + j] = v;
  }
}

__global__ void si_update(const float* __restrict__ imp, const float* __restrict__ si,
                          float* __restrict__ out) {
  int j = blockIdx.x * blockDim.x + threadIdx.x;
  if (j < Hc) {
    float s = 0.f;
    for (int b = 0; b < Bc; ++b) s += fabsf(imp[b * Hc + j]);
    s /= (float)Bc;
    out[j] = si[j] + 0.01f * (s - si[j]);
  }
}

// ---------------------------------------------------------------------------
// Launch pipeline
// ---------------------------------------------------------------------------
extern "C" void kernel_launch(void* const* d_in, const int* in_sizes, int n_in,
                              void* d_out, int out_size, void* d_ws, size_t ws_size,
                              hipStream_t stream) {
  (void)in_sizes; (void)n_in; (void)out_size; (void)ws_size;
  const float* x    = (const float*)d_in[0];
  const float* ctx  = (const float*)d_in[1];
  const float* si   = (const float*)d_in[2];
  const float* Wemb = (const float*)d_in[3];   const float* bemb = (const float*)d_in[4];
  const float* W0   = (const float*)d_in[5];   const float* b0   = (const float*)d_in[6];
  const float* g0   = (const float*)d_in[7];   const float* bt0  = (const float*)d_in[8];
  const float* W1   = (const float*)d_in[9];   const float* b1   = (const float*)d_in[10];
  const float* g1   = (const float*)d_in[11];  const float* bt1  = (const float*)d_in[12];
  const float* Wih0 = (const float*)d_in[13];  const float* Whh0 = (const float*)d_in[14];
  const float* bih0 = (const float*)d_in[15];  const float* bhh0 = (const float*)d_in[16];
  const float* Wih1 = (const float*)d_in[17];  const float* Whh1 = (const float*)d_in[18];
  const float* bih1 = (const float*)d_in[19];  const float* bhh1 = (const float*)d_in[20];
  const float* Wc   = (const float*)d_in[21];  const float* bc   = (const float*)d_in[22];
  const float* lncg = (const float*)d_in[23];  const float* lncb = (const float*)d_in[24];
  const float* Wi1  = (const float*)d_in[25];  const float* bi1  = (const float*)d_in[26];
  const float* Wi2  = (const float*)d_in[27];  const float* bi2  = (const float*)d_in[28];
  const float* lnig = (const float*)d_in[29];  const float* lnib = (const float*)d_in[30];
  const float* Wf   = (const float*)d_in[31];  const float* bf   = (const float*)d_in[32];
  float* out = (float*)d_out;

  // workspace carve (~410 MB)
  float* p = (float*)d_ws;
  float* hbuf1 = p; p += (size_t)Bc * Sc * Hc;
  float* hbuf2 = p; p += (size_t)Bc * Sc * Hc;
  float* X0    = p; p += (size_t)Bc * Sc * G4;
  float* c0    = p; p += Bc * Hc;          // zeroed region start (f32)
  float* c1    = p; p += Bc * Hc;
  float* h1f   = p; p += Bc * Hc;          // zeroed region end
  float* b0s   = p; p += G4;
  float* b1s   = p; p += G4;
  float* meanb = p; p += Sc;
  float* rvarb = p; p += Sc;
  float* catb  = p; p += Bc * 2 * Hc;
  float* tmpA  = p; p += Bc * Hc;
  float* comp  = p; p += Bc * Hc;
  float* cmpd  = p; p += Bc * Hc;
  float* t1    = p; p += Bc * Hc;
  float* t2    = p; p += Bc * Hc;
  float* imp   = p; p += Bc * Hc;
  __bf16* Whh0b = (__bf16*)p;
  __bf16* Wih1b = Whh0b + (size_t)G4 * Hc;
  __bf16* Whh1b = Wih1b + (size_t)G4 * Hc;
  __bf16* h0bf  = Whh1b + (size_t)G4 * Hc;   // zeroed int region start
  __bf16* h1bf  = h0bf + 2 * Bc * Hc;
  int* bar = (int*)(h1bf + 2 * Bc * Hc);     // 1024 counters (zeroed)

  const int nzf = 3 * Bc * Hc;                              // c0, c1, h1f
  const int nzi = (4 * Bc * Hc * 2 + 2 * Sc * 4) / 4;       // h0bf,h1bf,bar
  init_zero<<<512, BDIM, 0, stream>>>(c0, nzf, (int*)h0bf, nzi);

  const int wn = G4 * Hc;
  cvt_bf16<<<(wn + BDIM - 1) / BDIM, BDIM, 0, stream>>>(Whh0, Whh0b, wn);
  cvt_bf16<<<(wn + BDIM - 1) / BDIM, BDIM, 0, stream>>>(Wih1, Wih1b, wn);
  cvt_bf16<<<(wn + BDIM - 1) / BDIM, BDIM, 0, stream>>>(Whh1, Whh1b, wn);
  addvec<<<(G4 + BDIM - 1) / BDIM, BDIM, 0, stream>>>(bih0, bhh0, b0s, G4);
  addvec<<<(G4 + BDIM - 1) / BDIM, BDIM, 0, stream>>>(bih1, bhh1, b1s, G4);

  const int M = Bc * Sc;
  wmma_gemm_bias_act<<<dim3(M / 64, Hc / 128), BDIM, 0, stream>>>(
      x, Wemb, bemb, hbuf1, M, Hc, INc, 0);
  wmma_gemm_bias_act<<<dim3(M / 64, Hc / 128), BDIM, 0, stream>>>(
      hbuf1, W0, b0, hbuf2, M, Hc, Hc, 0);
  bn_stats<<<Sc, BDIM, 0, stream>>>(hbuf2, meanb, rvarb);
  bn_apply_relu<<<4096, BDIM, 0, stream>>>(hbuf2, meanb, rvarb, g0, bt0);
  wmma_gemm_bias_act<<<dim3(M / 64, Hc / 128), BDIM, 0, stream>>>(
      hbuf2, W1, b1, hbuf1, M, Hc, Hc, 0);
  bn_stats<<<Sc, BDIM, 0, stream>>>(hbuf1, meanb, rvarb);
  bn_apply_relu<<<4096, BDIM, 0, stream>>>(hbuf1, meanb, rvarb, g1, bt1);

  // hoisted layer-0 input GEMM: X0 = h @ Wih0^T + (bih0 + bhh0)
  wmma_gemm_bias_act<<<dim3(M / 64, G4 / 128), BDIM, 0, stream>>>(
      hbuf1, Wih0, b0s, X0, M, G4, Hc, 0);

  lstm_seq_kernel<<<8, BDIM, 0, stream>>>(X0, Whh0b, Wih1b, Whh1b, b1s,
                                          h0bf, c0, h1bf, c1, h1f, bar);

  concat_kernel<<<(Bc * 2 * Hc + BDIM - 1) / BDIM, BDIM, 0, stream>>>(h1f, ctx, catb);
  wmma_gemm_bias_act<<<dim3(1, Hc / 128), BDIM, 0, stream>>>(
      catb, Wc, bc, tmpA, Bc, Hc, 2 * Hc, 0);
  ln_act<<<Bc, BDIM, 0, stream>>>(tmpA, lncg, lncb, comp, Hc, 1);
  mul_kernel<<<(Bc * Hc + BDIM - 1) / BDIM, BDIM, 0, stream>>>(h1f, comp, cmpd, Bc * Hc);

  wmma_gemm_bias_act<<<dim3(1, Hc / 128), BDIM, 0, stream>>>(
      cmpd, Wi1, bi1, t1, Bc, Hc, Hc, 1);
  wmma_gemm_bias_act<<<dim3(1, Hc / 128), BDIM, 0, stream>>>(
      t1, Wi2, bi2, t2, Bc, Hc, Hc, 0);
  ln_act<<<Bc, BDIM, 0, stream>>>(t2, lnig, lnib, imp, Hc, 1);

  wmma_gemm_bias_act<<<dim3(1, Hc / 128), BDIM, 0, stream>>>(
      cmpd, Wf, bf, out, Bc, Hc, Hc, 0);
  si_update<<<(Hc + BDIM - 1) / BDIM, BDIM, 0, stream>>>(imp, si, out + Bc * Hc);
}
